// GT_FID_30391188587302
// MI455X (gfx1250) — matrix operand order repaired and checked by reading, not compiled, not run on hardware
//
#include <hip/hip_runtime.h>
#include <hip/hip_bf16.h>

typedef __attribute__((ext_vector_type(16))) __bf16 v16bf;
typedef __attribute__((ext_vector_type(8)))  __bf16 v8bf;
typedef __attribute__((ext_vector_type(8)))  float  v8f;

#define NWG_SCAN 16

__device__ __forceinline__ v8f wmma_bf16(v16bf a, v16bf b, v8f c) {
  return __builtin_amdgcn_wmma_f32_16x16x32_bf16(false, a, false, b, (short)0, c, false, false);
}
__device__ __forceinline__ __bf16 tobf(float f) { return (__bf16)f; }
__device__ __forceinline__ float sigm(float x) { return 1.f / (1.f + __expf(-x)); }

// A-fragment (16x32 bf16): lane half h holds contiguous K = 16h .. 16h+15 (matches B).
__device__ __forceinline__ v16bf load_afrag(const __bf16* __restrict__ row, int akb) {
  return *(const v16bf*)(row + 2 * akb);      // K = 2*akb .. 2*akb+15
}

// ---------------------------------------------------------------------------
// f32 -> packed bf16 conversion (one-time; keeps conversions out of hot loops)
// ---------------------------------------------------------------------------
__global__ void k_cvt_bf16(const float* __restrict__ in, __bf16* __restrict__ out, int n) {
  int i = blockIdx.x * 256 + threadIdx.x;
  if (i < n) out[i] = tobf(in[i]);
}

// ---------------------------------------------------------------------------
// GEMM: C[M,N] = A[M,K] * W[N,K]^T (+bias0+bias1). A,W packed bf16.
// M mult of 16, N mult of 32, K mult of 32. 16x32 output slab per wave:
// one A-fragment feeds two WMMAs (two N-tiles).
// ---------------------------------------------------------------------------
__global__ __launch_bounds__(256) void k_gemm_nt(
    const __bf16* __restrict__ A, const __bf16* __restrict__ W,
    const float* __restrict__ bias0, const float* __restrict__ bias1,
    float* __restrict__ C, int M, int N, int K, int hasBias) {
  const int wave = threadIdx.x >> 5;
  const int lane = threadIdx.x & 31;
  const int mt = M >> 4, nt2 = N >> 5;
  const int pair = blockIdx.x * 8 + wave;
  if (pair >= mt * nt2) return;               // wave-uniform exit; EXEC stays full
  const int tm = pair % mt;
  const int tn = (pair / mt) * 2;             // two adjacent N-tiles
  const int lhalf = lane >> 4;
  const int l15 = lane & 15;
  const int am  = (tm << 4) + l15;
  const int bn0 = (tn << 4) + l15;
  const int bn1 = bn0 + 16;
  const int akb = lhalf * 8;
  const int bkb = lhalf * 16;

  v8f acc0 = {}, acc1 = {};
  if (hasBias) {
    float bv0 = bias0[bn0] + (bias1 ? bias1[bn0] : 0.f);
    float bv1 = bias0[bn1] + (bias1 ? bias1[bn1] : 0.f);
#pragma unroll
    for (int v = 0; v < 8; v++) { acc0[v] = bv0; acc1[v] = bv1; }
  }
  const __bf16* Arow  = A + (size_t)am * K;
  const __bf16* Wrow0 = W + (size_t)bn0 * K + bkb;
  const __bf16* Wrow1 = W + (size_t)bn1 * K + bkb;
  for (int k0 = 0; k0 < K; k0 += 32) {
    v16bf af  = load_afrag(Arow + k0, akb);
    v16bf bf0 = *(const v16bf*)(Wrow0 + k0);
    v16bf bf1 = *(const v16bf*)(Wrow1 + k0);
    acc0 = wmma_bf16(af, bf0, acc0);
    acc1 = wmma_bf16(af, bf1, acc1);
  }
#pragma unroll
  for (int v = 0; v < 8; v++) {
    const int m = (tm << 4) + v + lhalf * 8;
    C[(size_t)m * N + bn0] = acc0[v];
    C[(size_t)m * N + bn1] = acc1[v];
  }
}

// ---------------------------------------------------------------------------
// Persistent LSTM scan: 16 WGs x 256 threads. WG k owns h columns [16k,16k+16).
// Per step: gates(64x64) = xp[t] + h @ Whh^T (bf16 WMMA; h kept in bf16).
// Wave's two M-tiles share one Whh B-fragment (WGP$-resident, 32KB/WG).
// Device-scope barrier each step (+ s_cluster_barrier, NOP when unclustered).
// ---------------------------------------------------------------------------
__device__ __forceinline__ void gbar(int* cnt, int* gen) {
  __syncthreads();
  if (threadIdx.x == 0) {
    __threadfence();
    atomicAdd(cnt, 1);
    const int target = NWG_SCAN * (*gen + 1);
    while (__hip_atomic_load(cnt, __ATOMIC_RELAXED, __HIP_MEMORY_SCOPE_AGENT) < target)
      __builtin_amdgcn_s_sleep(2);
  }
  __builtin_amdgcn_s_cluster_barrier();  // real cluster sync if dispatched clustered
  __syncthreads();
  __threadfence();
  (*gen)++;
}

__global__ __launch_bounds__(256) void k_lstm_scan(
    const float*  __restrict__ xp,    // [64,512,1024] (includes bih+bhh)
    const __bf16* __restrict__ WhhB,  // [1024,256] bf16
    const int*    __restrict__ lens,  // [64]
    __bf16*       __restrict__ h_bf,  // [64,256] bf16 (final h_lstm)
    int*          __restrict__ bar) {
  const int T = 512, H = 256, NG = 1024;
  const int wg = blockIdx.x;                 // 0..15
  const int tid = threadIdx.x;
  const int wave = tid >> 5, lane = tid & 31;
  const int lhalf = lane >> 4, l15 = lane & 15;
  __shared__ float g_lds[4][64][16];

  // Wave's work: gate g = wave>>1, M-tiles mt0, mt0+1
  const int g = wave >> 1;
  const int mt0 = (wave & 1) * 2;
  const int mt1 = mt0 + 1;
  const int gn = g * 256 + wg * 16 + l15;    // gate column (Whh row)
  const __bf16* wrow = WhhB + (size_t)gn * H + lhalf * 16;

  // Cell ownership: thread owns 4 cells (batch row cb, cols cj..cj+3 of WG's 16)
  const int cb = tid >> 2;
  const int cj = (tid & 3) * 4;
  const int mylen = lens[cb];
  float cst[4] = {0.f, 0.f, 0.f, 0.f};
#pragma unroll
  for (int i = 0; i < 4; i++) h_bf[cb * H + wg * 16 + cj + i] = tobf(0.f);
  int gen = 0;
  gbar(bar, &gen);

  for (int t = 0; t < T; t++) {
    v8f acc0, acc1;
#pragma unroll
    for (int v = 0; v < 8; v++) {
      const int m0 = mt0 * 16 + v + lhalf * 8;
      const int m1 = mt1 * 16 + v + lhalf * 8;
      const size_t col = (size_t)g * 256 + wg * 16 + l15;
      acc0[v] = xp[((size_t)m0 * T + t) * NG + col];
      acc1[v] = xp[((size_t)m1 * T + t) * NG + col];
    }
    for (int k0 = 0; k0 < H; k0 += 32) {
      v16bf bfr = *(const v16bf*)(wrow + k0);
      v16bf af0 = load_afrag(h_bf + (mt0 * 16 + l15) * H + k0, lhalf * 8);
      v16bf af1 = load_afrag(h_bf + (mt1 * 16 + l15) * H + k0, lhalf * 8);
      acc0 = wmma_bf16(af0, bfr, acc0);
      acc1 = wmma_bf16(af1, bfr, acc1);
    }
#pragma unroll
    for (int v = 0; v < 8; v++) {
      g_lds[g][mt0 * 16 + v + lhalf * 8][l15] = acc0[v];
      g_lds[g][mt1 * 16 + v + lhalf * 8][l15] = acc1[v];
    }
    __syncthreads();
    if (t < mylen) {
#pragma unroll
      for (int i = 0; i < 4; i++) {
        const int j = cj + i;
        const float gi = g_lds[0][cb][j], gf = g_lds[1][cb][j];
        const float gg = g_lds[2][cb][j], go = g_lds[3][cb][j];
        float cv = sigm(gf) * cst[i] + sigm(gi) * tanhf(gg);
        cst[i] = cv;
        h_bf[cb * H + wg * 16 + j] = tobf(sigm(go) * tanhf(cv));
      }
    }
    gbar(bar, &gen);
  }
}

// ---------------------------------------------------------------------------
// GCN helpers
// ---------------------------------------------------------------------------
__global__ void k_init_deg(float* deg, int N) {
  int i = blockIdx.x * 256 + threadIdx.x;
  if (i < N) deg[i] = 1.f;  // self loop
}
__global__ void k_deg_scatter(const int* __restrict__ dst, float* __restrict__ deg, int E) {
  int e = blockIdx.x * 256 + threadIdx.x;
  if (e < E) atomicAdd(&deg[dst[e]], 1.f);
}
__global__ void k_rsqrt_inplace(float* deg, int N) {
  int i = blockIdx.x * 256 + threadIdx.x;
  if (i < N) deg[i] = rsqrtf(deg[i]);
}
// Edge scatter: 4 edges per block, 64 lanes/edge, 4 features/lane.
// xw (51MB) + agg (51MB) are L2-resident (192MB L2) -> L2-atomic bound.
__global__ __launch_bounds__(256) void k_scatter(
    const float* __restrict__ xw, const float* __restrict__ dinv,
    const int* __restrict__ src, const int* __restrict__ dst,
    float* __restrict__ agg, int E) {
  const int e = blockIdx.x * 4 + (threadIdx.x >> 6);
  if (e >= E) return;
  const int s = src[e], d = dst[e];
  const float w = dinv[s] * dinv[d];
  const int f = (threadIdx.x & 63) * 4;
  const float4 v = *(const float4*)(xw + (size_t)s * 256 + f);
  float* o = agg + (size_t)d * 256 + f;
  atomicAdd(o + 0, v.x * w);
  atomicAdd(o + 1, v.y * w);
  atomicAdd(o + 2, v.z * w);
  atomicAdd(o + 3, v.w * w);
}
// hout = relu(agg + xw*dinv^2 + b) -> packed bf16 (next consumer is WMMA/pool)
__global__ void k_gcn_finish(const float* __restrict__ agg, const float* __restrict__ xw,
                             const float* __restrict__ dinv, const float* __restrict__ b,
                             __bf16* __restrict__ hout) {
  const int i = blockIdx.x, c = threadIdx.x;
  const float di = dinv[i];
  const size_t idx = (size_t)i * 256 + c;
  const float v = agg[idx] + xw[idx] * di * di + b[c];
  hout[idx] = tobf(v > 0.f ? v : 0.f);
}
__global__ void k_pool(const __bf16* __restrict__ h, const int* __restrict__ batch,
                       float* __restrict__ sums, float* __restrict__ cnt) {
  const int i = blockIdx.x, c = threadIdx.x;
  const int b = batch[i];
  atomicAdd(&sums[b * 256 + c], (float)h[(size_t)i * 256 + c]);
  if (c == 0) atomicAdd(&cnt[b], 1.f);
}

// ---------------------------------------------------------------------------
// Classifier (tiny)
// ---------------------------------------------------------------------------
__global__ void k_mlp1(const __bf16* __restrict__ hl, const float* __restrict__ sums,
                       const float* __restrict__ cnt, const float* __restrict__ Wc1,
                       const float* __restrict__ bc1, float* __restrict__ z) {
  const int b = blockIdx.x, j = threadIdx.x;  // grid 64, block 128
  const float inv = 1.f / fmaxf(cnt[b], 1.f);
  float acc = bc1[j];
  const float* wr = Wc1 + j * 512;
  for (int k = 0; k < 256; k++) acc += (float)hl[b * 256 + k] * wr[k];
  for (int k = 0; k < 256; k++) acc += sums[b * 256 + k] * inv * wr[256 + k];
  z[b * 128 + j] = fmaxf(acc, 0.f);
}
__global__ void k_mlp2(const float* __restrict__ z, const float* __restrict__ Wc2,
                       const float* __restrict__ bc2, float* __restrict__ out) {
  const int b = blockIdx.x, c = threadIdx.x;  // grid 64, block 2
  float acc = bc2[c];
  for (int j = 0; j < 128; j++) acc += z[b * 128 + j] * Wc2[c * 128 + j];
  out[b * 2 + c] = acc;
}

// ---------------------------------------------------------------------------
extern "C" void kernel_launch(void* const* d_in, const int* in_sizes, int n_in,
                              void* d_out, int out_size, void* d_ws, size_t ws_size,
                              hipStream_t stream) {
  const float* seqs = (const float*)d_in[0];
  const int*   lens = (const int*)d_in[1];
  const float* x    = (const float*)d_in[2];
  const int*   eidx = (const int*)d_in[3];
  const int*   batch= (const int*)d_in[4];
  const float* Wih  = (const float*)d_in[5];
  const float* Whh  = (const float*)d_in[6];
  const float* bih  = (const float*)d_in[7];
  const float* bhh  = (const float*)d_in[8];
  const float* W1   = (const float*)d_in[9];
  const float* b1   = (const float*)d_in[10];
  const float* W2   = (const float*)d_in[11];
  const float* b2   = (const float*)d_in[12];
  const float* Wc1  = (const float*)d_in[13];
  const float* bc1  = (const float*)d_in[14];
  const float* Wc2  = (const float*)d_in[15];
  const float* bc2  = (const float*)d_in[16];

  const int N = 50000, E = 1600000;
  const int* esrc = eidx;
  const int* edst = eidx + E;

  char* ws = (char*)d_ws;
  size_t o = 0;
  float*  xp     = (float*)(ws + o); o += (size_t)32768 * 1024 * 4;  // 134,217,728
  __bf16* h_bf   = (__bf16*)(ws + o); o += 64 * 256 * 2 + 128;       // 32,896 -> pad
  o = (o + 255) & ~(size_t)255;
  int*    bar    = (int*)(ws + o);    o += 256;
  float*  dinv   = (float*)(ws + o);  o += 204800;
  float*  xw     = (float*)(ws + o);  o += (size_t)N * 256 * 4;      // 51,200,000
  float*  agg    = (float*)(ws + o);  o += (size_t)N * 256 * 4;
  __bf16* h1_bf  = (__bf16*)(ws + o); o += (size_t)N * 256 * 2;      // 25,600,000
  __bf16* h2_bf  = (__bf16*)(ws + o); o += (size_t)N * 256 * 2;
  float*  sums   = (float*)(ws + o);  o += 64 * 256 * 4;
  float*  cnt    = (float*)(ws + o);  o += 256;
  float*  zbuf   = (float*)(ws + o);  o += 64 * 128 * 4;
  __bf16* seqsB  = (__bf16*)(ws + o); o += (size_t)64 * 512 * 128 * 2;  // 8,388,608
  __bf16* WihB   = (__bf16*)(ws + o); o += 1024 * 128 * 2;
  __bf16* WhhB   = (__bf16*)(ws + o); o += 1024 * 256 * 2;
  __bf16* W1B    = (__bf16*)(ws + o); o += 256 * 256 * 2;
  __bf16* W2B    = (__bf16*)(ws + o); o += 256 * 256 * 2;
  // xp region reuse (only after the scan has consumed xp):
  __bf16* x_bf   = (__bf16*)((char*)xp);               // 25.6 MB
  float*  agg2   = (float*)((char*)xp + 33554432);     // 51.2 MB, disjoint from x_bf

  float* outp = (float*)d_out;

  // ---- one-time bf16 packing of LSTM operands ----
  hipMemsetAsync(bar, 0, 256, stream);  // deterministic across graph replays
  {
    int n;
    n = 64 * 512 * 128; k_cvt_bf16<<<(n + 255) / 256, 256, 0, stream>>>(seqs, seqsB, n);
    n = 1024 * 128;     k_cvt_bf16<<<(n + 255) / 256, 256, 0, stream>>>(Wih, WihB, n);
    n = 1024 * 256;     k_cvt_bf16<<<(n + 255) / 256, 256, 0, stream>>>(Whh, WhhB, n);
    n = 256 * 256;      k_cvt_bf16<<<(n + 255) / 256, 256, 0, stream>>>(W1, W1B, n);
    n = 256 * 256;      k_cvt_bf16<<<(n + 255) / 256, 256, 0, stream>>>(W2, W2B, n);
  }

  // ---- LSTM ----
  {
    const int pairs = (32768 / 16) * (1024 / 32);  // 65536
    k_gemm_nt<<<(pairs + 7) / 8, 256, 0, stream>>>(seqsB, WihB, bih, bhh, xp,
                                                   32768, 1024, 128, 1);
  }
  k_lstm_scan<<<NWG_SCAN, 256, 0, stream>>>(xp, WhhB, lens, h_bf, bar);

  // ---- GCN degree ----
  k_init_deg<<<(N + 255) / 256, 256, 0, stream>>>(dinv, N);
  k_deg_scatter<<<(E + 255) / 256, 256, 0, stream>>>(edst, dinv, E);
  k_rsqrt_inplace<<<(N + 255) / 256, 256, 0, stream>>>(dinv, N);

  // ---- GCN layer 1 ----
  {
    int n = N * 256;  // pack x after the scan (x_bf aliases the xp region)
    k_cvt_bf16<<<(n + 255) / 256, 256, 0, stream>>>(x, x_bf, n);
    const int pairs = (N / 16) * (256 / 32);  // 25000
    k_gemm_nt<<<(pairs + 7) / 8, 256, 0, stream>>>(x_bf, W1B, nullptr, nullptr, xw,
                                                   N, 256, 256, 0);
  }
  hipMemsetAsync(agg, 0, (size_t)N * 256 * 4, stream);
  k_scatter<<<(E + 3) / 4, 256, 0, stream>>>(xw, dinv, esrc, edst, agg, E);
  k_gcn_finish<<<N, 256, 0, stream>>>(agg, xw, dinv, b1, h1_bf);

  // ---- GCN layer 2 ----
  {
    const int pairs = (N / 16) * (256 / 32);
    k_gemm_nt<<<(pairs + 7) / 8, 256, 0, stream>>>(h1_bf, W2B, nullptr, nullptr, xw,
                                                   N, 256, 256, 0);
  }
  hipMemsetAsync(agg2, 0, (size_t)N * 256 * 4, stream);
  k_scatter<<<(E + 3) / 4, 256, 0, stream>>>(xw, dinv, esrc, edst, agg2, E);
  k_gcn_finish<<<N, 256, 0, stream>>>(agg2, xw, dinv, b2, h2_bf);

  // ---- pool + classifier ----
  hipMemsetAsync(sums, 0, 64 * 256 * 4 + 256, stream);  // sums + cnt contiguous
  k_pool<<<N, 256, 0, stream>>>(h2_bf, batch, sums, cnt);
  k_mlp1<<<64, 128, 0, stream>>>(h_bf, sums, cnt, Wc1, bc1, zbuf);
  k_mlp2<<<64, 2, 0, stream>>>(zbuf, Wc2, bc2, outp);
}
